// FP8StorageLinear_10943576670523
// MI455X (gfx1250) — matrix operand order repaired and checked by reading, not compiled, not run on hardware
//
#include <hip/hip_runtime.h>
#include <hip/hip_bf16.h>

typedef __attribute__((ext_vector_type(16))) __bf16 v16bf;
typedef __attribute__((ext_vector_type(8)))  __bf16 v8bf;
typedef __attribute__((ext_vector_type(8)))  float  v8f;
typedef __attribute__((ext_vector_type(4)))  int    v4i_;

#define M_DIM 16384
#define N_DIM 11008
#define K_DIM 4096
#define BM 128
#define BN 256
#define BK 64
#define LDW 72   // padded LDS stride in halves (144B = 9*16B: aligned, conflict-free)
#define NTHREADS 256
#define NKT (K_DIM / BK)

#if __has_builtin(__builtin_amdgcn_global_load_async_to_lds_b128)
#define USE_ASYNC 1
#else
#define USE_ASYNC 0
#endif

// ---------------- fp8 e4m3 -> f32 (handles denorm + NaN) ----------------
__device__ __forceinline__ float e4m3_to_f32(unsigned b) {
  unsigned e = (b >> 3) & 0xFu;
  unsigned m = b & 7u;
  float v;
  if (e == 0u) {
    v = (float)m * 0.001953125f;                          // m * 2^-9
  } else {
    v = __uint_as_float(((e + 120u) << 23) | (m << 20));  // rebias 7 -> 127
  }
  if ((b & 0x7Fu) == 0x7Fu) v = __uint_as_float(0x7FC00000u);  // e4m3fn NaN
  return (b & 0x80u) ? -v : v;
}

__device__ __forceinline__ unsigned short f32_to_bf16_bits(float f) {
  __hip_bfloat16 h = __float2bfloat16(f);
  unsigned short u;
  __builtin_memcpy(&u, &h, 2);
  return u;
}

// ---------------- Kernel 1: dequantize W fp8 -> bf16*scale into d_ws ----------------
__global__ __launch_bounds__(256) void fp8lin_dequant_kernel(
    const unsigned char* __restrict__ w8, const float* __restrict__ scale,
    unsigned short* __restrict__ wb) {
  const size_t base = ((size_t)blockIdx.x * 256u + threadIdx.x) * 8u;
  const float s = scale[0];
  uint2 p = *(const uint2*)(w8 + base);
  union { unsigned short u16[8]; uint4 v; } o;
#pragma unroll
  for (int i = 0; i < 4; ++i)
    o.u16[i] = f32_to_bf16_bits(e4m3_to_f32((p.x >> (8 * i)) & 0xFFu) * s);
#pragma unroll
  for (int i = 0; i < 4; ++i)
    o.u16[4 + i] = f32_to_bf16_bits(e4m3_to_f32((p.y >> (8 * i)) & 0xFFu) * s);
  *(uint4*)(wb + base) = o.v;
}

// ---------------- async / staging helpers ----------------
#if USE_ASYNC
typedef __attribute__((address_space(1))) v4i_* gv4p;  // global int4*
typedef __attribute__((address_space(3))) v4i_* lv4p;  // LDS int4*

__device__ __forceinline__ void async_ld16(const unsigned short* g,
                                           unsigned short* l) {
  __builtin_amdgcn_global_load_async_to_lds_b128((gv4p)(unsigned short*)g,
                                                 (lv4p)l, 0, 0);
}

template <int N>
__device__ __forceinline__ void wait_asynccnt() {
#if __has_builtin(__builtin_amdgcn_s_wait_asynccnt)
  __builtin_amdgcn_s_wait_asynccnt(N);
#else
  asm volatile("s_wait_asynccnt %0" ::"n"(N) : "memory");
#endif
}

// 12 async b128 loads per thread per k-tile: 4 for A (128x64), 8 for B (256x64)
__device__ __forceinline__ void issue_tile_async(
    const unsigned short* __restrict__ X, const unsigned short* __restrict__ Wb,
    unsigned short* a_s, unsigned short* b_s, int mBase, int nBase, int kBase,
    int t) {
  const int col8 = (t & 7) << 3;
  const int row0 = t >> 3;
#pragma unroll
  for (int p = 0; p < 4; ++p) {
    const int r = row0 + (p << 5);
    async_ld16(X + (size_t)(mBase + r) * K_DIM + kBase + col8,
               a_s + r * LDW + col8);
  }
#pragma unroll
  for (int p = 0; p < 8; ++p) {
    const int r = row0 + (p << 5);
    async_ld16(Wb + (size_t)(nBase + r) * K_DIM + kBase + col8,
               b_s + r * LDW + col8);
  }
}
#else
__device__ __forceinline__ void g2r(const unsigned short* __restrict__ gA,
                                    const unsigned short* __restrict__ gB,
                                    int mBase, int nBase, int kBase, int t,
                                    uint4* ra, uint4* rb) {
  const int col8 = (t & 7) << 3;
  const int row0 = t >> 3;
#pragma unroll
  for (int p = 0; p < 4; ++p)
    ra[p] = *(const uint4*)(gA + (size_t)(mBase + row0 + (p << 5)) * K_DIM +
                            kBase + col8);
#pragma unroll
  for (int p = 0; p < 8; ++p)
    rb[p] = *(const uint4*)(gB + (size_t)(nBase + row0 + (p << 5)) * K_DIM +
                            kBase + col8);
}

__device__ __forceinline__ void r2lds(unsigned short* a_s, unsigned short* b_s,
                                      int t, const uint4* ra, const uint4* rb) {
  const int col8 = (t & 7) << 3;
  const int row0 = t >> 3;
#pragma unroll
  for (int p = 0; p < 4; ++p)
    *(uint4*)(a_s + (row0 + (p << 5)) * LDW + col8) = ra[p];
#pragma unroll
  for (int p = 0; p < 8; ++p)
    *(uint4*)(b_s + (row0 + (p << 5)) * LDW + col8) = rb[p];
}
#endif

// ---------------- Kernel 2: bf16 WMMA GEMM  out = X * Wb^T + bias ----------------
// 256 threads = 8 waves as 2(M) x 4(N); wave tile 64x64 = 4x4 WMMA accumulators.
__global__ __launch_bounds__(NTHREADS) void fp8lin_gemm_kernel(
    const unsigned short* __restrict__ X,   // [M,K] bf16 bits
    const unsigned short* __restrict__ Wb,  // [N,K] bf16 bits (dequantized)
    const __hip_bfloat16* __restrict__ bias,
    unsigned short* __restrict__ out) {     // [M,N] bf16 bits
  __shared__ unsigned short sA[2][BM * LDW];  //  36 KB
  __shared__ unsigned short sB[2][BN * LDW];  //  72 KB

  // Grouped rasterization for L2 reuse (GROUP_M = 8)
  const int num_pid_m = M_DIM / BM;  // 128
  const int num_pid_n = N_DIM / BN;  // 43
  const int pid = blockIdx.x;
  const int GROUPM = 8;
  const int npg = GROUPM * num_pid_n;
  const int gid = pid / npg;
  const int first_m = gid * GROUPM;
  const int gsz = (num_pid_m - first_m) < GROUPM ? (num_pid_m - first_m) : GROUPM;
  const int pid_m = first_m + (pid % gsz);
  const int pid_n = (pid % npg) / gsz;

  const int mBase = pid_m * BM;
  const int nBase = pid_n * BN;
  const int t = threadIdx.x;
  const int lane = t & 31;
  const int wave = t >> 5;
  const int wm = wave >> 2;  // 0..1 : 64 rows of M
  const int wn = wave & 3;   // 0..3 : 64 cols of N

  v8f acc[4][4];
  const v8f vzero = {0.f, 0.f, 0.f, 0.f, 0.f, 0.f, 0.f, 0.f};
#pragma unroll
  for (int mi = 0; mi < 4; ++mi)
#pragma unroll
    for (int ni = 0; ni < 4; ++ni) acc[mi][ni] = vzero;

#if USE_ASYNC
  issue_tile_async(X, Wb, sA[0], sB[0], mBase, nBase, 0, t);
#else
  uint4 ra[4], rb[8];
  g2r(X, Wb, mBase, nBase, 0, t, ra, rb);
  r2lds(sA[0], sB[0], t, ra, rb);
  __syncthreads();
#endif

  for (int kt = 0; kt < NKT; ++kt) {
    const int cur = kt & 1;
#if USE_ASYNC
    if (kt + 1 < NKT) {
      issue_tile_async(X, Wb, sA[cur ^ 1], sB[cur ^ 1], mBase, nBase,
                       (kt + 1) * BK, t);
      wait_asynccnt<12>();  // batch kt complete (in-order), kt+1 in flight
    } else {
      wait_asynccnt<0>();
    }
    __syncthreads();
#else
    if (kt + 1 < NKT) g2r(X, Wb, mBase, nBase, (kt + 1) * BK, t, ra, rb);
#endif

    const unsigned short* a_s = sA[cur];
    const unsigned short* b_s = sB[cur];
#pragma unroll
    for (int kc = 0; kc < BK; kc += 32) {
      v16bf afr[4];
      v16bf bfr[4];
      // A (16x32): lanes 0-15 row m, K kc+[0..7]/[16..23];
      //            lanes 16-31 same rows, K kc+[8..15]/[24..31]
#pragma unroll
      for (int mi = 0; mi < 4; ++mi) {
        const int row = wm * 64 + mi * 16 + (lane & 15);
        const int k1 = kc + ((lane >> 4) << 3);
        v8bf lo = *(const v8bf*)(a_s + row * LDW + k1);
        v8bf hi = *(const v8bf*)(a_s + row * LDW + k1 + 16);
        afr[mi] = __builtin_shufflevector(lo, hi, 0, 1, 2, 3, 4, 5, 6, 7, 8, 9,
                                          10, 11, 12, 13, 14, 15);
      }
      // B (32x16, B = W^T): column N = lane&15; lanes 0-15 K kc+[0..7]/[16..23],
      //                     lanes 16-31 K kc+[8..15]/[24..31]
#pragma unroll
      for (int ni = 0; ni < 4; ++ni) {
        const int row = wn * 64 + ni * 16 + (lane & 15);  // W row = N index
        const int k2 = kc + ((lane >> 4) << 3);
        v8bf lo = *(const v8bf*)(b_s + row * LDW + k2);
        v8bf hi = *(const v8bf*)(b_s + row * LDW + k2 + 16);
        bfr[ni] = __builtin_shufflevector(lo, hi, 0, 1, 2, 3, 4, 5, 6, 7, 8, 9,
                                          10, 11, 12, 13, 14, 15);
      }
#pragma unroll
      for (int mi = 0; mi < 4; ++mi)
#pragma unroll
        for (int ni = 0; ni < 4; ++ni)
          acc[mi][ni] = __builtin_amdgcn_wmma_f32_16x16x32_bf16(
              false, afr[mi], false, bfr[ni], (short)0, acc[mi][ni], false,
              false);
    }

#if USE_ASYNC
    __syncthreads();  // everyone done reading buf cur before kt+1 writes it
#else
    if (kt + 1 < NKT) r2lds(sA[cur ^ 1], sB[cur ^ 1], t, ra, rb);
    __syncthreads();
#endif
  }

  // Epilogue: C/D layout -> VGPR j holds M=j (lanes 0-15) / M=8+j (lanes 16-31)
  const int cN = lane & 15;
  const int rsel = (lane >> 4) << 3;
#pragma unroll
  for (int ni = 0; ni < 4; ++ni) {
    const int col = nBase + wn * 64 + ni * 16 + cN;
    const float bf = __bfloat162float(bias[col]);
#pragma unroll
    for (int mi = 0; mi < 4; ++mi) {
      const int rbase = mBase + wm * 64 + mi * 16 + rsel;
#pragma unroll
      for (int j = 0; j < 8; ++j) {
        const unsigned short bits = f32_to_bf16_bits(acc[mi][ni][j] + bf);
        // stream out non-temporally: keep X + Wb resident in L2
        __builtin_nontemporal_store(bits,
                                    out + (size_t)(rbase + j) * N_DIM + col);
      }
    }
  }
}

// ---------------- Launch ----------------
extern "C" void kernel_launch(void* const* d_in, const int* in_sizes, int n_in,
                              void* d_out, int out_size, void* d_ws, size_t ws_size,
                              hipStream_t stream) {
  const unsigned short* X = (const unsigned short*)d_in[0];     // bf16 [M,K]
  const unsigned char* W8 = (const unsigned char*)d_in[1];      // fp8  [N,K]
  const float* scale = (const float*)d_in[2];                   // f32  [1]
  const __hip_bfloat16* bias = (const __hip_bfloat16*)d_in[3];  // bf16 [N]
  unsigned short* Wb = (unsigned short*)d_ws;                   // bf16 [N,K]
  unsigned short* out = (unsigned short*)d_out;                 // bf16 [M,N]

  // 1) dequantize weights once: 11008*4096 = 45,088,768 elems, 8/thread
  const int dq_blocks = (N_DIM * K_DIM) / (8 * 256);  // 22016
  fp8lin_dequant_kernel<<<dq_blocks, 256, 0, stream>>>(W8, scale, Wb);

  // 2) WMMA bf16 GEMM over 128x256 tiles
  const int gemm_blocks = (M_DIM / BM) * (N_DIM / BN);  // 128*43 = 5504
  fp8lin_gemm_kernel<<<gemm_blocks, NTHREADS, 0, stream>>>(X, Wb, bias, out);
}
